// CNNVariateMixing_55061480735202
// MI455X (gfx1250) — matrix operand order repaired and checked by reading, not compiled, not run on hardware
//
#include <hip/hip_runtime.h>
#include <hip/hip_bf16.h>

typedef __attribute__((ext_vector_type(16))) _Float16 v16h;
typedef __attribute__((ext_vector_type(4)))  _Float16 v4h;
typedef __attribute__((ext_vector_type(8)))  float    v8f;
typedef __attribute__((ext_vector_type(4)))  int      v4i;

#define D_DIM   1024
#define W_WIN   10
#define PADW    12
#define NWIN    1536            // B * (T/W) = 16 * 96
#define M_TOTAL 15360           // NWIN * W_WIN
#define K1      3072            // 3 * D
#define K2      1024
#define BM      128
#define BN      128
#define BK      32

// ---------------------------------------------------------------------------
// CDNA5 async global->LDS copy (ASYNCcnt path), with safe fallback.
// Builtin signature (from hipcc diagnostic): first param is int4* in the
// global segment; cast both pointers to exact typed vector pointers.
// ---------------------------------------------------------------------------
#if defined(__HIP_DEVICE_COMPILE__) && defined(__gfx1250__) && \
    __has_builtin(__builtin_amdgcn_global_load_async_to_lds_b128)
#define HAVE_ASYNC_LDS 1
#else
#define HAVE_ASYNC_LDS 0
#endif

#if HAVE_ASYNC_LDS
typedef __attribute__((address_space(1))) v4i gas_v4i;
typedef __attribute__((address_space(3))) v4i las_v4i;
#endif

__device__ __forceinline__ void async_cp16(const _Float16* __restrict__ g,
                                           _Float16* __restrict__ l) {
#if HAVE_ASYNC_LDS
    __builtin_amdgcn_global_load_async_to_lds_b128(
        (gas_v4i*)g, (las_v4i*)l, 0 /*offset*/, 0 /*cpol*/);
#else
    *(uint4*)l = *(const uint4*)g;
#endif
}

__device__ __forceinline__ void wait_async_lds() {
#if HAVE_ASYNC_LDS
#if __has_builtin(__builtin_amdgcn_s_wait_asynccnt)
    __builtin_amdgcn_s_wait_asynccnt(0);
#else
    asm volatile("s_wait_asynccnt 0x0" ::: "memory");
#endif
#endif
}

// ---------------------------------------------------------------------------
// Pack kernels: fp32 -> f16, weight re-layout
// ---------------------------------------------------------------------------
__global__ void pack_x_kernel(const float* __restrict__ x,
                              _Float16* __restrict__ xpad) {
    // xpad: [NWIN][12][1024]; rows 0 and 11 are the zero padding,
    // row r (1..10) = x row (wi*10 + r - 1). 4 elements per thread.
    long t = (long)blockIdx.x * blockDim.x + threadIdx.x;
    long e = t * 4;
    const long TOT = (long)NWIN * PADW * D_DIM;
    if (e >= TOT) return;
    int  c  = (int)(e & (D_DIM - 1));
    int  r  = (int)((e >> 10) % PADW);
    long wi = e / ((long)PADW * D_DIM);
    v4h h = {};
    if (r != 0 && r != PADW - 1) {
        const float4 f = *(const float4*)(x + (wi * W_WIN + (r - 1)) * D_DIM + c);
        h[0] = (_Float16)f.x; h[1] = (_Float16)f.y;
        h[2] = (_Float16)f.z; h[3] = (_Float16)f.w;
    }
    *(v4h*)(xpad + e) = h;
}

__global__ void pack_convw_kernel(const float* __restrict__ conv_w,
                                  _Float16* __restrict__ cw) {
    // cw[d][kk] with kk = k*1024 + c  <-  conv_w[d][c][k] (stride-3 gather)
    int idx = blockIdx.x * blockDim.x + threadIdx.x;
    if (idx >= D_DIM * K1) return;
    int d  = idx / K1;
    int kk = idx - d * K1;
    int k  = kk >> 10;
    int c  = kk & 1023;
    cw[idx] = (_Float16)conv_w[d * K1 + c * 3 + k];
}

__global__ void pack_gatew_kernel(const float* __restrict__ gate_w,
                                  _Float16* __restrict__ gw) {
    long t = (long)blockIdx.x * blockDim.x + threadIdx.x;
    long e = t * 4;
    if (e >= (long)D_DIM * D_DIM) return;
    const float4 f = *(const float4*)(gate_w + e);
    v4h h;
    h[0] = (_Float16)f.x; h[1] = (_Float16)f.y;
    h[2] = (_Float16)f.z; h[3] = (_Float16)f.w;
    *(v4h*)(gw + e) = h;
}

// ---------------------------------------------------------------------------
// GEMM 1: cnn[m][d] = (f16) sum_kk xpad_row(m)[kk] * cw[d][kk] + conv_b[d]
//   A rows are overlapping contiguous 3072-elem slices of xpad.
// ---------------------------------------------------------------------------
__global__ __launch_bounds__(256)
void conv_gemm_kernel(const _Float16* __restrict__ Axp,
                      const _Float16* __restrict__ Bw,
                      const float*    __restrict__ bias,
                      _Float16*       __restrict__ Cout) {
    __shared__ __align__(16) _Float16 As[2][BM * BK];
    __shared__ __align__(16) _Float16 Bs[2][BN * BK];

    const int tid   = threadIdx.x;
    const int lane  = tid & 31;
    const int wave  = tid >> 5;
    const int waveM = wave & 1;     // 2 wave-rows  * 64 M
    const int waveN = wave >> 1;    // 4 wave-cols  * 32 N
    const int bm    = blockIdx.x * BM;
    const int bn    = blockIdx.y * BN;

    // loader mapping: thread -> rows (lr, lr+64), 16B chunk column lc
    const int lr = tid >> 2;
    const int lc = (tid & 3) << 3;

    long aBase0, aBase1;
    {
        int m0 = bm + lr;
        int m1 = bm + lr + 64;
        aBase0 = ((long)(m0 / W_WIN) * PADW + (m0 % W_WIN)) * D_DIM;
        aBase1 = ((long)(m1 / W_WIN) * PADW + (m1 % W_WIN)) * D_DIM;
    }
    const long bBase0 = (long)(bn + lr) * K1;
    const long bBase1 = (long)(bn + lr + 64) * K1;

    auto stage = [&](int buf, long ko) {
        async_cp16(Axp + aBase0 + ko, &As[buf][lr * BK + lc]);
        async_cp16(Axp + aBase1 + ko, &As[buf][(lr + 64) * BK + lc]);
        async_cp16(Bw  + bBase0 + ko, &Bs[buf][lr * BK + lc]);
        async_cp16(Bw  + bBase1 + ko, &Bs[buf][(lr + 64) * BK + lc]);
    };

    v8f acc[4][2] = {};

    stage(0, lc);
    wait_async_lds();
    __syncthreads();

    const int KT   = K1 / BK;
    const int klo  = (lane < 16) ? 0 : 8;   // ISA A/B fragment K striping
    const int arow = waveM * 64 + (lane & 15);
    const int brow = waveN * 32 + (lane & 15);

    for (int kt = 0; kt < KT; ++kt) {
        const int cur = kt & 1;
        if (kt + 1 < KT) stage((kt + 1) & 1, (long)(kt + 1) * BK + lc);

        v16h bfrag[2];
#pragma unroll
        for (int j = 0; j < 2; ++j) {
            const _Float16* p = &Bs[cur][(brow + j * 16) * BK + klo];
            ((uint4*)&bfrag[j])[0] = *(const uint4*)p;
            ((uint4*)&bfrag[j])[1] = *(const uint4*)(p + 16);
        }
#pragma unroll
        for (int i = 0; i < 4; ++i) {
            v16h afrag;
            const _Float16* p = &As[cur][(arow + i * 16) * BK + klo];
            ((uint4*)&afrag)[0] = *(const uint4*)p;
            ((uint4*)&afrag)[1] = *(const uint4*)(p + 16);
#pragma unroll
            for (int j = 0; j < 2; ++j) {
                acc[i][j] = __builtin_amdgcn_wmma_f32_16x16x32_f16(
                    false, afrag, false, bfrag[j], (short)0, acc[i][j],
                    false, false);
            }
        }

        if (kt + 1 < KT) {
            wait_async_lds();
            __syncthreads();
        }
    }

    // epilogue: + bias, f16 store.  C/D layout: lane -> col, VGPR r -> row
#pragma unroll
    for (int j = 0; j < 2; ++j) {
        const int   ncol = bn + waveN * 32 + j * 16 + (lane & 15);
        const float bv   = bias[ncol];
#pragma unroll
        for (int i = 0; i < 4; ++i) {
            const int mrow = bm + waveM * 64 + i * 16 + ((lane >> 4) << 3);
#pragma unroll
            for (int r = 0; r < 8; ++r) {
                Cout[(long)(mrow + r) * D_DIM + ncol] =
                    (_Float16)(acc[i][j][r] + bv);
            }
        }
    }
}

// ---------------------------------------------------------------------------
// GEMM 2: g = cnn @ gate_w^T + gate_b ; out = cnn * sigmoid(g)
//   A (cnn) is f16 in scratch; both tiles staged via async LDS copies.
// ---------------------------------------------------------------------------
__global__ __launch_bounds__(256)
void gate_gemm_kernel(const _Float16* __restrict__ cnnh,
                      const _Float16* __restrict__ Bw,
                      const float*    __restrict__ bias,
                      float*          __restrict__ Out) {
    __shared__ __align__(16) _Float16 As[2][BM * BK];
    __shared__ __align__(16) _Float16 Bs[2][BN * BK];

    const int tid   = threadIdx.x;
    const int lane  = tid & 31;
    const int wave  = tid >> 5;
    const int waveM = wave & 1;
    const int waveN = wave >> 1;
    const int bm    = blockIdx.x * BM;
    const int bn    = blockIdx.y * BN;

    const int lr = tid >> 2;
    const int lc = (tid & 3) << 3;

    const long aBase0 = (long)(bm + lr) * K2;
    const long aBase1 = (long)(bm + lr + 64) * K2;
    const long bBase0 = (long)(bn + lr) * K2;
    const long bBase1 = (long)(bn + lr + 64) * K2;

    auto stage = [&](int buf, long ko) {
        async_cp16(cnnh + aBase0 + ko, &As[buf][lr * BK + lc]);
        async_cp16(cnnh + aBase1 + ko, &As[buf][(lr + 64) * BK + lc]);
        async_cp16(Bw   + bBase0 + ko, &Bs[buf][lr * BK + lc]);
        async_cp16(Bw   + bBase1 + ko, &Bs[buf][(lr + 64) * BK + lc]);
    };

    v8f acc[4][2] = {};

    stage(0, lc);
    wait_async_lds();
    __syncthreads();

    const int KT   = K2 / BK;
    const int klo  = (lane < 16) ? 0 : 8;
    const int arow = waveM * 64 + (lane & 15);
    const int brow = waveN * 32 + (lane & 15);

    for (int kt = 0; kt < KT; ++kt) {
        const int cur = kt & 1;
        if (kt + 1 < KT) stage((kt + 1) & 1, (long)(kt + 1) * BK + lc);

        v16h bfrag[2];
#pragma unroll
        for (int j = 0; j < 2; ++j) {
            const _Float16* p = &Bs[cur][(brow + j * 16) * BK + klo];
            ((uint4*)&bfrag[j])[0] = *(const uint4*)p;
            ((uint4*)&bfrag[j])[1] = *(const uint4*)(p + 16);
        }
#pragma unroll
        for (int i = 0; i < 4; ++i) {
            v16h afrag;
            const _Float16* p = &As[cur][(arow + i * 16) * BK + klo];
            ((uint4*)&afrag)[0] = *(const uint4*)p;
            ((uint4*)&afrag)[1] = *(const uint4*)(p + 16);
#pragma unroll
            for (int j = 0; j < 2; ++j) {
                acc[i][j] = __builtin_amdgcn_wmma_f32_16x16x32_f16(
                    false, afrag, false, bfrag[j], (short)0, acc[i][j],
                    false, false);
            }
        }

        if (kt + 1 < KT) {
            wait_async_lds();
            __syncthreads();
        }
    }

    // epilogue: out = cnn * sigmoid(acc + gate_b)
#pragma unroll
    for (int j = 0; j < 2; ++j) {
        const int   ncol = bn + waveN * 32 + j * 16 + (lane & 15);
        const float bv   = bias[ncol];
#pragma unroll
        for (int i = 0; i < 4; ++i) {
            const int mrow = bm + waveM * 64 + i * 16 + ((lane >> 4) << 3);
#pragma unroll
            for (int r = 0; r < 8; ++r) {
                const long idx = (long)(mrow + r) * D_DIM + ncol;
                const float g  = acc[i][j][r] + bv;
                const float s  = 1.0f / (1.0f + __expf(-g));
                Out[idx] = (float)cnnh[idx] * s;
            }
        }
    }
}

// ---------------------------------------------------------------------------
// Host launcher
// ---------------------------------------------------------------------------
extern "C" void kernel_launch(void* const* d_in, const int* in_sizes, int n_in,
                              void* d_out, int out_size, void* d_ws, size_t ws_size,
                              hipStream_t stream) {
    const float* x      = (const float*)d_in[0];
    const float* conv_w = (const float*)d_in[1];
    const float* conv_b = (const float*)d_in[2];
    const float* gate_w = (const float*)d_in[3];
    const float* gate_b = (const float*)d_in[4];
    float*       out    = (float*)d_out;

    // workspace layout (all 256B-aligned by construction); total ~74 MB
    char* ws = (char*)d_ws;
    const size_t XPAD_B = (size_t)NWIN * PADW * D_DIM * sizeof(_Float16); // 37,748,736
    const size_t CW_B   = (size_t)D_DIM * K1 * sizeof(_Float16);          //  6,291,456
    const size_t GW_B   = (size_t)D_DIM * D_DIM * sizeof(_Float16);       //  2,097,152
    _Float16* xpad = (_Float16*)ws;
    _Float16* cw   = (_Float16*)(ws + XPAD_B);
    _Float16* gw   = (_Float16*)(ws + XPAD_B + CW_B);
    _Float16* cnnh = (_Float16*)(ws + XPAD_B + CW_B + GW_B);              // 31,457,280

    // pack / convert (fp32 -> f16, conv weight re-layout)
    {
        const long nx = (long)NWIN * PADW * D_DIM / 4;          // 4,718,592
        pack_x_kernel<<<(unsigned)((nx + 255) / 256), 256, 0, stream>>>(x, xpad);
    }
    {
        const int ncw = D_DIM * K1;                              // 3,145,728
        pack_convw_kernel<<<(ncw + 255) / 256, 256, 0, stream>>>(conv_w, cw);
    }
    {
        const int ngw = D_DIM * D_DIM / 4;                       // 262,144
        pack_gatew_kernel<<<(ngw + 255) / 256, 256, 0, stream>>>(gate_w, gw);
    }

    // GEMM 1: conv -> cnn (f16 scratch)
    conv_gemm_kernel<<<dim3(M_TOTAL / BM, D_DIM / BN), 256, 0, stream>>>(
        xpad, cw, conv_b, cnnh);

    // GEMM 2: gate + fused sigmoid/multiply -> out
    gate_gemm_kernel<<<dim3(M_TOTAL / BM, D_DIM / BN), 256, 0, stream>>>(
        cnnh, gw, gate_b, out);
}
